// MPNN_25194278158451
// MI455X (gfx1250) — matrix operand order repaired and checked by reading, not compiled, not run on hardware
//
#include <hip/hip_runtime.h>
#include <hip/hip_bf16.h>

// ---------------------------------------------------------------------------
// MPNN node update for MI455X (gfx1250, wave32, WMMA bf16 16x16x32)
// ---------------------------------------------------------------------------

#define HID   64
#define H3    192
#define MID   128
#define TILE_NODES_PER_BLOCK 128   // 8 waves * 16 rows
#define NTHREADS 256

typedef __attribute__((ext_vector_type(16))) __bf16 bf16x16;
typedef __attribute__((ext_vector_type(8)))  __bf16 bf16x8;
typedef __attribute__((ext_vector_type(8)))  float  f32x8;

// A-matrix fragment (16-bit, 16x32): lane half h holds contiguous K block
// {kbase+16h .. kbase+16h+15}  (two 16B loads)
__device__ __forceinline__ bf16x16 load_a_frag(const __bf16* base, int rowStride,
                                               int row, int kc0) {
  const __bf16* p = base + row * rowStride + kc0;
  union { bf16x16 v; bf16x8 h[2]; } u;
  u.h[0] = *(const bf16x8*)(p + (kc0 & 8));
  u.h[1] = *(const bf16x8*)(p + (kc0 & 8) + 8);
  return u.v;
}

// B-matrix fragment (16-bit, 32x16): lane half h holds contiguous K block
// {kbase+16h .. kbase+16h+15} of column n  (one 32B load, Wt is [Nout][K])
__device__ __forceinline__ bf16x16 load_b_frag(const __bf16* Wt, int kStride,
                                               int col, int kbase, int half) {
  return *(const bf16x16*)(Wt + col * kStride + kbase + half * 16);
}

// ---------------------------------------------------------------------------
// grep[f] = sum over nodes of hidden[:, f]
// ---------------------------------------------------------------------------
__global__ void __launch_bounds__(NTHREADS)
mpnn_reduce_grep(const float* __restrict__ hin, float* __restrict__ grep, int nNodes) {
  __shared__ float sbuf[NTHREADS];
  int tid = threadIdx.x;
  int f = tid & 63;
  int g = tid >> 6;                       // 4 node-groups per block
  float s = 0.f;
  for (int node = blockIdx.x * 4 + g; node < nNodes; node += gridDim.x * 4)
    s += hin[(size_t)node * HID + f];
  sbuf[tid] = s;
  __syncthreads();
  if (g == 0) {
    float t = sbuf[tid] + sbuf[tid + 64] + sbuf[tid + 128] + sbuf[tid + 192];
    atomicAdd(&grep[f], t);
  }
}

// ---------------------------------------------------------------------------
// agg[dst[e], :] += hidden[src[e], :]   (64 threads per edge, coalesced;
// src/dst are wave-uniform -> scalar loads)
// ---------------------------------------------------------------------------
__global__ void __launch_bounds__(NTHREADS)
mpnn_scatter(const float* __restrict__ hin, const int* __restrict__ src,
             const int* __restrict__ dst, float* __restrict__ agg, int nEdges) {
  int tid = threadIdx.x;
  int f = tid & 63;
  int e = blockIdx.x * 4 + (tid >> 6);
  if (e < nEdges) {
    int s = src[e];
    int d = dst[e];
    float v = hin[(size_t)s * HID + f];
    atomicAdd(&agg[(size_t)d * HID + f], v);
  }
}

// ---------------------------------------------------------------------------
// Fused MLP: x=[agg|hid|grep] -> relu(xW1+b1) -> relu(hW2+b2) -> hW3+b3 -> L2norm
// One 256-thread block = 8 waves; each wave owns a 16-row tile.
// All weights (bf16, transposed [Nout][K]) cached in LDS.
// ---------------------------------------------------------------------------
__global__ void __launch_bounds__(NTHREADS, 1)
mpnn_mlp(const float* __restrict__ agg, const float* __restrict__ hin,
         const float* __restrict__ grep,
         const float* __restrict__ W1, const float* __restrict__ b1,
         const float* __restrict__ W2, const float* __restrict__ b2,
         const float* __restrict__ W3, const float* __restrict__ b3,
         float* __restrict__ hout, int nNodes) {
  extern __shared__ char smem[];
  __bf16* Wt1 = (__bf16*)smem;                 // [128][192] bf16  49152 B
  __bf16* Wt2 = Wt1 + MID * H3;                // [128][128] bf16  32768 B
  __bf16* Wt3 = Wt2 + MID * MID;               // [ 64][128] bf16  16384 B
  float*  bs1 = (float*)(Wt3 + HID * MID);     // 128 f32
  float*  bs2 = bs1 + MID;                     // 128 f32
  float*  bs3 = bs2 + MID;                     //  64 f32
  __bf16* xbase = (__bf16*)(bs3 + HID);        // 8 * 16*192 bf16  49152 B
  __bf16* hbase = xbase + 8 * 16 * H3;         // 8 * 16*128 bf16  32768 B
  float*  rnorm = (float*)(hbase + 8 * 16 * MID); // 8 * 16 f32

  int tid = threadIdx.x;

  // --- cooperative weight load: fp32 global (coalesced) -> bf16 LDS transposed
  for (int i = tid; i < MID * H3; i += NTHREADS) {
    int n = i & (MID - 1), k = i >> 7;         // i % 128, i / 128
    Wt1[n * H3 + k] = (__bf16)W1[k * MID + n];
  }
  for (int i = tid; i < MID * MID; i += NTHREADS) {
    int n = i & (MID - 1), k = i >> 7;
    Wt2[n * MID + k] = (__bf16)W2[k * MID + n];
  }
  for (int i = tid; i < HID * MID; i += NTHREADS) {
    int n = i & (HID - 1), k = i >> 6;
    Wt3[n * MID + k] = (__bf16)W3[k * HID + n];
  }
  if (tid < MID) bs1[tid] = b1[tid];
  if (tid < MID) bs2[tid] = b2[tid];
  if (tid < HID) bs3[tid] = b3[tid];
  __syncthreads();

  const int wave  = tid >> 5;
  const int lane  = tid & 31;
  const int half  = lane >> 4;
  const int l16   = lane & 15;
  const int node0 = blockIdx.x * TILE_NODES_PER_BLOCK + wave * 16;

  __bf16* xw = xbase + wave * (16 * H3);       // x tile  [16][192] bf16
  __bf16* hw = hbase + wave * (16 * MID);      // h1 tile [16][128] bf16
  float*  ow = (float*)hw;                     // out tile [16][64] f32 (aliases hw)
  float*  rw = rnorm + wave * 16;

  // --- build x = [agg | hidden | grep] as bf16
  for (int i = lane; i < 16 * H3; i += 32) {
    int r = i / H3, c = i - r * H3;
    int node = node0 + r;
    float v = 0.f;
    if (node < nNodes) {
      if (c < HID)           v = agg[(size_t)node * HID + c];
      else if (c < 2 * HID)  v = hin[(size_t)node * HID + (c - HID)];
      else                   v = grep[c - 2 * HID];
    }
    xw[i] = (__bf16)v;
  }
  asm volatile("s_wait_dscnt 0" ::: "memory");

  // --- layer 1: [16x192] @ [192x128], K-steps of 32
  f32x8 acc[8];
#pragma unroll
  for (int n = 0; n < 8; ++n) acc[n] = (f32x8)0.0f;
  for (int ks = 0; ks < 6; ++ks) {
    int kbase = ks * 32;
    bf16x16 a = load_a_frag(xw, H3, l16, kbase + half * 8);
#pragma unroll
    for (int n = 0; n < 8; ++n) {
      bf16x16 b = load_b_frag(Wt1, H3, n * 16 + l16, kbase, half);
      acc[n] = __builtin_amdgcn_wmma_f32_16x16x32_bf16(
          false, a, false, b, (short)0, acc[n], false, false);
    }
  }
#pragma unroll
  for (int n = 0; n < 8; ++n) {              // bias + relu -> h1 bf16
    int col = n * 16 + l16;
    float bb = bs1[col];
#pragma unroll
    for (int j = 0; j < 8; ++j) {
      int m = half * 8 + j;
      float v = acc[n][j] + bb;
      hw[m * MID + col] = (__bf16)(v > 0.f ? v : 0.f);
    }
  }
  asm volatile("s_wait_dscnt 0" ::: "memory");

  // --- layer 2: [16x128] @ [128x128] -> h2 (reuse x region)
#pragma unroll
  for (int n = 0; n < 8; ++n) acc[n] = (f32x8)0.0f;
  for (int ks = 0; ks < 4; ++ks) {
    int kbase = ks * 32;
    bf16x16 a = load_a_frag(hw, MID, l16, kbase + half * 8);
#pragma unroll
    for (int n = 0; n < 8; ++n) {
      bf16x16 b = load_b_frag(Wt2, MID, n * 16 + l16, kbase, half);
      acc[n] = __builtin_amdgcn_wmma_f32_16x16x32_bf16(
          false, a, false, b, (short)0, acc[n], false, false);
    }
  }
  __bf16* h2 = xw;                            // x fully consumed -> reuse
#pragma unroll
  for (int n = 0; n < 8; ++n) {
    int col = n * 16 + l16;
    float bb = bs2[col];
#pragma unroll
    for (int j = 0; j < 8; ++j) {
      int m = half * 8 + j;
      float v = acc[n][j] + bb;
      h2[m * MID + col] = (__bf16)(v > 0.f ? v : 0.f);
    }
  }
  asm volatile("s_wait_dscnt 0" ::: "memory");

  // --- layer 3: [16x128] @ [128x64] -> out f32 (aliases h1 region)
  f32x8 acc3[4];
#pragma unroll
  for (int n = 0; n < 4; ++n) acc3[n] = (f32x8)0.0f;
  for (int ks = 0; ks < 4; ++ks) {
    int kbase = ks * 32;
    bf16x16 a = load_a_frag(h2, MID, l16, kbase + half * 8);
#pragma unroll
    for (int n = 0; n < 4; ++n) {
      bf16x16 b = load_b_frag(Wt3, MID, n * 16 + l16, kbase, half);
      acc3[n] = __builtin_amdgcn_wmma_f32_16x16x32_bf16(
          false, a, false, b, (short)0, acc3[n], false, false);
    }
  }
#pragma unroll
  for (int n = 0; n < 4; ++n) {
    int col = n * 16 + l16;
    float bb = bs3[col];
#pragma unroll
    for (int j = 0; j < 8; ++j) {
      int m = half * 8 + j;
      ow[m * HID + col] = acc3[n][j] + bb;
    }
  }
  asm volatile("s_wait_dscnt 0" ::: "memory");

  // --- row-wise L2 norm
  if (lane < 16) {
    float s = 0.f;
#pragma unroll
    for (int c = 0; c < HID; ++c) {
      float v = ow[lane * HID + c];
      s += v * v;
    }
    rw[lane] = (s > 0.f) ? (1.f / sqrtf(s)) : 0.f;
  }
  asm volatile("s_wait_dscnt 0" ::: "memory");

  for (int i = lane; i < 16 * HID; i += 32) {
    int r = i >> 6, c = i & 63;
    int node = node0 + r;
    if (node < nNodes)
      hout[(size_t)node * HID + c] = ow[i] * rw[r];
  }
}

// LDS: Wt1 49152 + Wt2 32768 + Wt3 16384 + biases 1280 + x 49152 + h 32768 + rn 512
static const size_t MLP_SMEM = 49152 + 32768 + 16384 + 1280 + 49152 + 32768 + 512;

// ---------------------------------------------------------------------------
extern "C" void kernel_launch(void* const* d_in, const int* in_sizes, int n_in,
                              void* d_out, int out_size, void* d_ws, size_t ws_size,
                              hipStream_t stream) {
  const float* hidden = (const float*)d_in[0];
  const int*   src    = (const int*)d_in[1];
  const int*   dst    = (const int*)d_in[2];
  const float* W1     = (const float*)d_in[3];
  const float* b1     = (const float*)d_in[4];
  const float* W2     = (const float*)d_in[5];
  const float* b2     = (const float*)d_in[6];
  const float* W3     = (const float*)d_in[7];
  const float* b3     = (const float*)d_in[8];
  float* out = (float*)d_out;

  const int N = in_sizes[0] / HID;
  const int E = in_sizes[1];

  float* ws   = (float*)d_ws;
  float* agg  = ws;                          // N*64
  float* grep = agg + (size_t)N * HID;       // 64 (contiguous with agg for one memset)
  float* bufA = grep + HID;                  // N*64
  float* bufB = bufA + (size_t)N * HID;      // N*64

  (void)hipFuncSetAttribute((const void*)mpnn_mlp,
                            hipFuncAttributeMaxDynamicSharedMemorySize,
                            (int)MLP_SMEM);

  const float* ins[4]  = {hidden, bufA, bufB, bufA};
  float*       outs[4] = {bufA,   bufB, bufA, out};

  const int gridMlp     = (N + TILE_NODES_PER_BLOCK - 1) / TILE_NODES_PER_BLOCK;
  const int gridScatter = (E + 3) / 4;

  for (int step = 0; step < 4; ++step) {
    const int blk = step >> 1;               // 2 iters per block
    const float* hin  = ins[step];
    float*       hout = outs[step];

    hipMemsetAsync(agg, 0, ((size_t)N * HID + HID) * sizeof(float), stream);

    mpnn_reduce_grep<<<512, NTHREADS, 0, stream>>>(hin, grep, N);

    mpnn_scatter<<<gridScatter, NTHREADS, 0, stream>>>(hin, src, dst, agg, E);

    mpnn_mlp<<<gridMlp, NTHREADS, MLP_SMEM, stream>>>(
        agg, hin, grep,
        W1 + (size_t)blk * H3  * MID, b1 + (size_t)blk * MID,
        W2 + (size_t)blk * MID * MID, b2 + (size_t)blk * MID,
        W3 + (size_t)blk * MID * HID, b3 + (size_t)blk * HID,
        hout, N);
  }
}
